// MoELayer_85710367359162
// MI455X (gfx1250) — matrix operand order repaired and checked by reading, not compiled, not run on hardware
//
#include <hip/hip_runtime.h>
#include <hip/hip_bf16.h>
#include <stdint.h>

typedef __attribute__((ext_vector_type(16))) __bf16 v16bf;
typedef __attribute__((ext_vector_type(8)))  float  v8f;

#define Bb 4
#define Ss 2048
#define Dd 512
#define Ff 2048
#define Ee 8
#define Tt (Bb * Ss)      // 8192 tokens
#define BM 64             // tokens per FFN block (4 WMMA M-tiles)
#define XSTRIDE 520       // x row stride in LDS: 1040B, 16B-aligned, bank-skewed
#define HSTRIDE 136       // h row stride in LDS: 272B, 16B-aligned, bank-skewed

union FragBF {
  v16bf bf;
  uint4 u[2];
};

// ---------------------------------------------------------------- utilities

__global__ void zero_cnt_kernel(int* cnt) {
  if (threadIdx.x < Ee) cnt[threadIdx.x] = 0;
}

__global__ void cvt_x_kernel(const float* __restrict__ x, __bf16* __restrict__ xb, int n) {
  int i = blockIdx.x * blockDim.x + threadIdx.x;
  if (i < n) xb[i] = (__bf16)x[i];
}

// src: [E][R][C] f32 row-major  ->  dst: [E][C][R] bf16 row-major
__global__ __launch_bounds__(256) void transpose_cvt_kernel(
    const float* __restrict__ src, __bf16* __restrict__ dst, int R, int C) {
  __shared__ float tile[32][33];
  const int e  = blockIdx.z;
  const int c0 = blockIdx.x * 32;
  const int r0 = blockIdx.y * 32;
  const float* s = src + (size_t)e * R * C;
  __bf16*      d = dst + (size_t)e * R * C;
  const int tx = threadIdx.x & 31;
  const int ty = threadIdx.x >> 5;   // 0..7
#pragma unroll
  for (int i = 0; i < 32; i += 8)
    tile[ty + i][tx] = s[(size_t)(r0 + ty + i) * C + (c0 + tx)];
  __syncthreads();
#pragma unroll
  for (int i = 0; i < 32; i += 8)
    d[(size_t)(c0 + ty + i) * R + (r0 + tx)] = (__bf16)tile[tx][ty + i];
}

__global__ void combine_kernel(const float* __restrict__ outp, float* __restrict__ out, int n) {
  int i = blockIdx.x * blockDim.x + threadIdx.x;
  if (i < n) out[i] = outp[i] + outp[(size_t)n + i];
}

// ---------------------------------------------------------------- router

__global__ __launch_bounds__(256) void router_kernel(
    const float* __restrict__ x, const float* __restrict__ Wr, const float* __restrict__ br,
    int* __restrict__ cnt, int* __restrict__ btok,
    float* __restrict__ bgate, int* __restrict__ bslot) {
  const int lane = threadIdx.x & 31;
  const int t = (int)((blockIdx.x * blockDim.x + threadIdx.x) >> 5);
  if (t >= Tt) return;
  const float* xr = x + (size_t)t * Dd;

  float acc[Ee];
#pragma unroll
  for (int e = 0; e < Ee; ++e) acc[e] = 0.f;
  for (int d = lane; d < Dd; d += 32) {
    const float xv = xr[d];
    const float* w = Wr + (size_t)d * Ee;
#pragma unroll
    for (int e = 0; e < Ee; ++e) acc[e] += xv * w[e];
  }
#pragma unroll
  for (int off = 16; off > 0; off >>= 1) {
#pragma unroll
    for (int e = 0; e < Ee; ++e) acc[e] += __shfl_xor(acc[e], off, 32);
  }
#pragma unroll
  for (int e = 0; e < Ee; ++e) acc[e] += br[e];

  float mx = acc[0];
#pragma unroll
  for (int e = 1; e < Ee; ++e) mx = fmaxf(mx, acc[e]);
  float p[Ee];
#pragma unroll
  for (int e = 0; e < Ee; ++e) p[e] = __expf(acc[e] - mx);
  int i1 = 0;
#pragma unroll
  for (int e = 1; e < Ee; ++e) if (p[e] > p[i1]) i1 = e;
  int i2 = (i1 == 0) ? 1 : 0;
#pragma unroll
  for (int e = 0; e < Ee; ++e) if (e != i1 && p[e] > p[i2]) i2 = e;
  const float inv = 1.f / (p[i1] + p[i2]);

  if (lane == 0) {
    int s0 = atomicAdd(&cnt[i1], 1);
    btok [(size_t)i1 * Tt + s0] = t;
    bgate[(size_t)i1 * Tt + s0] = p[i1] * inv;
    bslot[(size_t)i1 * Tt + s0] = 0;
    int s1 = atomicAdd(&cnt[i2], 1);
    btok [(size_t)i2 * Tt + s1] = t;
    bgate[(size_t)i2 * Tt + s1] = p[i2] * inv;
    bslot[(size_t)i2 * Tt + s1] = 1;
  }
}

// ---------------------------------------------------------------- fused expert FFN
// Block: 64 tokens (4 M-tiles) of expert blockIdx.y, 256 threads = 8 waves.
// Phase 1: wave w computes h cols [w*16, w*16+16) for all 4 M-tiles (B reuse x4).
// Phase 2: wave w owns out cols [w*64, w*64+64) for all 4 M-tiles (A & B reuse x4).
// s_h is double-buffered so each round needs only ONE workgroup barrier.

__global__ __launch_bounds__(256) void moe_ffn_kernel(
    const __bf16* __restrict__ xb,    // [T][D] bf16
    const __bf16* __restrict__ W1t,   // [E][F][D] bf16 (transposed)
    const __bf16* __restrict__ W2t,   // [E][D][F] bf16 (transposed)
    const float* __restrict__ b1,     // [E][F]
    const float* __restrict__ b2,     // [E][D]
    const int* __restrict__ cnt,
    const int* __restrict__ btok, const float* __restrict__ bgate,
    const int* __restrict__ bslot,
    float* __restrict__ outp)         // [2][T][D]
{
  const int e  = blockIdx.y;
  const int m0 = blockIdx.x * BM;
  const int nc = cnt[e];
  if (m0 >= nc) return;

  __shared__ int   s_tok[BM];
  __shared__ float s_gate[BM];
  __shared__ int   s_slot[BM];
  __shared__ __align__(16) __bf16 s_x[BM * XSTRIDE];
  __shared__ __align__(16) __bf16 s_h[2][BM * HSTRIDE];

  const int tid  = threadIdx.x;
  const int lane = tid & 31;
  const int w    = tid >> 5;

  if (tid < BM) {
    const int i = m0 + tid;
    const bool valid = i < nc;
    const int src = valid ? i : m0;
    s_tok [tid] = btok[(size_t)e * Tt + src];
    s_gate[tid] = valid ? bgate[(size_t)e * Tt + src] : 0.f;
    s_slot[tid] = valid ? bslot[(size_t)e * Tt + src] : -1;
  }
  __syncthreads();

  // stage x rows into LDS: 4 threads per row, 128 elements (256B) each
  {
    const int row  = tid >> 2;   // 0..63
    const int part = tid & 3;    // 0..3
    const __bf16* src = xb + (size_t)s_tok[row] * Dd + part * 128;
    __bf16* dst = &s_x[row * XSTRIDE + part * 128];
#pragma unroll
    for (int j = 0; j < 16; ++j)
      *(uint4*)(dst + j * 8) = *(const uint4*)(src + j * 8);
  }
  __syncthreads();

  const int nlo = lane & 15;
  const int hi  = lane >> 4;

  const __bf16* w1b = W1t + ((size_t)e * Ff + (size_t)(w * 16 + nlo)) * Dd;
  const __bf16* w2b = W2t + ((size_t)e * Dd + (size_t)(w * 64 + nlo)) * Ff;

  v8f acc[4][4];   // [m][nt]
#pragma unroll
  for (int m = 0; m < 4; ++m)
#pragma unroll
    for (int nt = 0; nt < 4; ++nt)
#pragma unroll
      for (int i = 0; i < 8; ++i) acc[m][nt][i] = 0.f;

#pragma unroll 1
  for (int r = 0; r < 16; ++r) {
    const int f0 = r * 128;
    __bf16* hbuf = &s_h[r & 1][0];

    // warm next round's weight streams (global_prefetch_b8 on gfx1250)
    {
      const int f1 = (r < 15) ? f0 + 128 : f0;
      __builtin_prefetch(w1b + (size_t)f1 * Dd, 0, 1);
      __builtin_prefetch(w2b + f1, 0, 1);
    }

    // -------- phase 1: h[64 x 16@w] = relu(x @ W1 + b1)
    v8f hc[4];
#pragma unroll
    for (int m = 0; m < 4; ++m)
#pragma unroll
      for (int i = 0; i < 8; ++i) hc[m][i] = 0.f;

    const __bf16* w1p = w1b + (size_t)f0 * Dd;
#pragma unroll
    for (int k0 = 0; k0 < Dd; k0 += 32) {
      // batch all fragment loads for this k-step, then issue 4 WMMAs
      FragBF b, a[4];
      const __bf16* bp = w1p + k0 + hi * 8;
      b.u[0] = *(const uint4*)(bp);
      b.u[1] = *(const uint4*)(bp + 16);
#pragma unroll
      for (int m = 0; m < 4; ++m) {
        const __bf16* ap = &s_x[(m * 16 + nlo) * XSTRIDE + k0 + hi * 8];
        a[m].u[0] = *(const uint4*)(ap);
        a[m].u[1] = *(const uint4*)(ap + 16);
      }
#pragma unroll
      for (int m = 0; m < 4; ++m)
        hc[m] = __builtin_amdgcn_wmma_f32_16x16x32_bf16(false, a[m].bf, false, b.bf,
                                                        (short)0, hc[m], false, false);
    }
    {
      const float bias = b1[(size_t)e * Ff + f0 + w * 16 + nlo];
#pragma unroll
      for (int m = 0; m < 4; ++m)
#pragma unroll
        for (int i = 0; i < 8; ++i) {
          float v = hc[m][i] + bias;
          v = v > 0.f ? v : 0.f;
          hbuf[(m * 16 + i + 8 * hi) * HSTRIDE + (w * 16 + nlo)] = (__bf16)v;
        }
    }
    __syncthreads();   // single barrier per round (s_h double-buffered)

    // -------- phase 2: acc[64 x 64@w] += h[64 x 128] @ W2 chunk
#pragma unroll
    for (int k0 = 0; k0 < 128; k0 += 32) {
      // batch all 8 fragment loads for this k-step, then issue 16 WMMAs
      FragBF a[4], b[4];
#pragma unroll
      for (int m = 0; m < 4; ++m) {
        const __bf16* ap = &hbuf[(m * 16 + nlo) * HSTRIDE + k0 + hi * 8];
        a[m].u[0] = *(const uint4*)(ap);
        a[m].u[1] = *(const uint4*)(ap + 16);
      }
#pragma unroll
      for (int nt = 0; nt < 4; ++nt) {
        const __bf16* bp = w2b + (size_t)nt * 16 * Ff + f0 + k0 + hi * 8;
        b[nt].u[0] = *(const uint4*)(bp);
        b[nt].u[1] = *(const uint4*)(bp + 16);
      }
#pragma unroll
      for (int nt = 0; nt < 4; ++nt)
#pragma unroll
        for (int m = 0; m < 4; ++m)
          acc[m][nt] = __builtin_amdgcn_wmma_f32_16x16x32_bf16(false, a[m].bf, false, b[nt].bf,
                                                               (short)0, acc[m][nt], false, false);
    }
  }

  // -------- epilogue: (+b2) * gate, scatter to out_partial[slot][tok][d]
#pragma unroll
  for (int nt = 0; nt < 4; ++nt) {
    const int d = w * 64 + nt * 16 + nlo;
    const float bias2 = b2[(size_t)e * Dd + d];
#pragma unroll
    for (int m = 0; m < 4; ++m)
#pragma unroll
      for (int i = 0; i < 8; ++i) {
        const int row  = m * 16 + i + 8 * hi;
        const int slot = s_slot[row];
        if (slot >= 0) {
          const float v = (acc[m][nt][i] + bias2) * s_gate[row];
          outp[((size_t)slot * Tt + s_tok[row]) * Dd + d] = v;
        }
      }
  }
}

// ---------------------------------------------------------------- launch

extern "C" void kernel_launch(void* const* d_in, const int* in_sizes, int n_in,
                              void* d_out, int out_size, void* d_ws, size_t ws_size,
                              hipStream_t stream) {
  (void)in_sizes; (void)n_in; (void)out_size; (void)ws_size;
  const float* x  = (const float*)d_in[0];
  const float* Wr = (const float*)d_in[1];
  const float* br = (const float*)d_in[2];
  const float* W1 = (const float*)d_in[3];
  const float* b1 = (const float*)d_in[4];
  const float* W2 = (const float*)d_in[5];
  const float* b2 = (const float*)d_in[6];
  float* out = (float*)d_out;

  char* ws = (char*)d_ws;
  size_t off = 0;
  auto alloc = [&](size_t bytes) -> void* {
    off = (off + 255) & ~(size_t)255;
    void* p = ws + off;
    off += bytes;
    return p;
  };
  __bf16* xb   = (__bf16*)alloc((size_t)Tt * Dd * 2);
  __bf16* W1t  = (__bf16*)alloc((size_t)Ee * Ff * Dd * 2);
  __bf16* W2t  = (__bf16*)alloc((size_t)Ee * Dd * Ff * 2);
  int*    btok = (int*)  alloc((size_t)Ee * Tt * 4);
  float*  bgate= (float*)alloc((size_t)Ee * Tt * 4);
  int*    bslot= (int*)  alloc((size_t)Ee * Tt * 4);
  int*    cnt  = (int*)  alloc(256);
  float*  outp = (float*)alloc((size_t)2 * Tt * Dd * 4);

  zero_cnt_kernel<<<1, 32, 0, stream>>>(cnt);
  cvt_x_kernel<<<(Tt * Dd + 255) / 256, 256, 0, stream>>>(x, xb, Tt * Dd);
  transpose_cvt_kernel<<<dim3(Ff / 32, Dd / 32, Ee), 256, 0, stream>>>(W1, W1t, Dd, Ff);
  transpose_cvt_kernel<<<dim3(Dd / 32, Ff / 32, Ee), 256, 0, stream>>>(W2, W2t, Ff, Dd);
  router_kernel<<<Tt / 8, 256, 0, stream>>>(x, Wr, br, cnt, btok, bgate, bslot);
  moe_ffn_kernel<<<dim3(Tt / BM, Ee), 256, 0, stream>>>(xb, W1t, W2t, b1, b2,
                                                        cnt, btok, bgate, bslot, outp);
  combine_kernel<<<(Tt * Dd + 255) / 256, 256, 0, stream>>>(outp, out, Tt * Dd);
}
